// GCN_8409545965927
// MI455X (gfx1250) — matrix operand order repaired and checked by reading, not compiled, not run on hardware
//
#include <hip/hip_runtime.h>
#include <hip/hip_bf16.h>
#include <math.h>

typedef __attribute__((ext_vector_type(16))) _Float16 v16h;
typedef __attribute__((ext_vector_type(8)))  float    v8f;

// ---------------------------------------------------------------------------
// Fragment loaders matching CDNA5 16-bit WMMA VGPR layouts (CDNA5 ISA)
// A (16x32, MxK), tile stored row-major [row][stride]:
//   lane<16:  row=lane,    elems 0..7 = K0..7,  elems 8..15 = K16..23
//   lane>=16: row=lane-16, elems 0..7 = K8..15, elems 8..15 = K24..31
__device__ __forceinline__ v16h load_frag_a(const _Float16* base, int stride, int lane) {
    const _Float16* p = base + (lane & 15) * stride + ((lane >> 4) << 3);
    v16h a;
#pragma unroll
    for (int i = 0; i < 8; ++i) { a[i] = p[i]; a[i + 8] = p[i + 16]; }
    return a;
}
// B (32x16, KxN) with tile stored N-major [col][stride] (pre-transposed weights):
//   lane<16:  col=lane,    elems 0..7 = K0..7,  elems 8..15 = K16..23
//   lane>=16: col=lane-16, elems 0..7 = K8..15, elems 8..15 = K24..31
__device__ __forceinline__ v16h load_frag_b(const _Float16* base, int stride, int lane) {
    const _Float16* p = base + (lane & 15) * stride + ((lane >> 4) << 3);
    v16h b;
#pragma unroll
    for (int i = 0; i < 8; ++i) { b[i] = p[i]; b[i + 8] = p[i + 16]; }
    return b;
}

// ---------------------------------------------------------------------------
// Small utility kernels
__global__ void fill_f32(float* __restrict__ p, float v, size_t n) {
    size_t i = (size_t)blockIdx.x * 256 + threadIdx.x;
    if (i < n) p[i] = v;
}

__global__ void deg_accum(const int* __restrict__ dst, float* __restrict__ deg, int E) {
    int e = blockIdx.x * 256 + threadIdx.x;
    if (e < E) atomicAdd(&deg[dst[e]], 1.0f);
}

__global__ void deg_to_dinv(float* __restrict__ deg, int M) {
    int i = blockIdx.x * 256 + threadIdx.x;
    if (i < M) deg[i] = rsqrtf(deg[i]);   // self-loop guarantees deg >= 1
}

// W1 [500][128] f32 -> W1t [128][512] f16 (zero-padded K)
__global__ void prep_w1t(const float* __restrict__ W1, _Float16* __restrict__ w1t) {
    int i = blockIdx.x * 256 + threadIdx.x;
    if (i >= 128 * 512) return;
    int n = i >> 9, k = i & 511;
    w1t[i] = (k < 500) ? (_Float16)W1[(size_t)k * 128 + n] : (_Float16)0.f;
}

// W2 [128][40] f32 -> W2t [48][128] f16 (zero-padded N)
__global__ void prep_w2t(const float* __restrict__ W2, _Float16* __restrict__ w2t) {
    int i = blockIdx.x * 256 + threadIdx.x;
    if (i >= 48 * 128) return;
    int n = i >> 7, k = i & 127;
    w2t[i] = (n < 40) ? (_Float16)W2[(size_t)k * 40 + n] : (_Float16)0.f;
}

// ---------------------------------------------------------------------------
// GEMM1: t1[M][128] = x[M][500] @ W1[500][128], f16 WMMA, fp32 accumulate.
// Block: 256 thr (8 waves), tile M=128 x N=128, K padded to 512 (16 steps of 32).
__global__ __launch_bounds__(256) void gemm1_wmma(const float* __restrict__ x,
                                                  const _Float16* __restrict__ w1t,
                                                  float* __restrict__ t1, int M) {
    __shared__ __align__(16) _Float16 As[128 * 32];   // 8 KB
    __shared__ __align__(16) _Float16 Bs[128 * 32];   // 8 KB (N-major)
    const int tid  = threadIdx.x;
    const int lane = tid & 31;
    const int wave = tid >> 5;
    const int wm = wave >> 1;           // 0..3 -> M offset wm*32
    const int wn = wave & 1;            // 0..1 -> N offset wn*64
    const int block_m = blockIdx.x * 128;

    v8f acc[2][4] = {};

    const int srow = tid >> 1;          // 0..127
    const int sc0  = (tid & 1) * 16;    // 0 or 16
    const int gr   = block_m + srow;
    const bool rowOK = (gr < M);
    const float* xrow = x + (size_t)gr * 500;

    for (int kk = 0; kk < 16; ++kk) {
        __syncthreads();
        const int gk = kk * 32 + sc0;
        // ---- stage A tile: chunk-granular guard, vector loads, b128 stores
        {
            __align__(16) _Float16 tmp[16];
            if (rowOK && (gk + 16) <= 500) {
                const float4* xp4 = (const float4*)(xrow + gk);
#pragma unroll
                for (int q = 0; q < 4; ++q) {
                    const float4 f = xp4[q];
                    tmp[q * 4 + 0] = (_Float16)f.x;
                    tmp[q * 4 + 1] = (_Float16)f.y;
                    tmp[q * 4 + 2] = (_Float16)f.z;
                    tmp[q * 4 + 3] = (_Float16)f.w;
                }
            } else {
#pragma unroll
                for (int i = 0; i < 16; ++i) {
                    float v = 0.f;
                    if (rowOK && (gk + i) < 500) v = xrow[gk + i];
                    tmp[i] = (_Float16)v;
                }
            }
            float4* ap4 = (float4*)(As + srow * 32 + sc0);
            ap4[0] = *(const float4*)(tmp);
            ap4[1] = *(const float4*)(tmp + 8);

            // ---- stage B tile: padded f16 weights, straight 16B copies
            const float4* wp4 = (const float4*)(w1t + (size_t)srow * 512 + kk * 32 + sc0);
            float4* bp4 = (float4*)(Bs + srow * 32 + sc0);
            bp4[0] = wp4[0];
            bp4[1] = wp4[1];

            // prefetch next K chunk of x (global_prefetch_b8)
            if (rowOK && kk < 15) __builtin_prefetch(xrow + gk + 32, 0, 1);
        }
        __syncthreads();

        v16h afrag[2], bfrag[4];
#pragma unroll
        for (int i = 0; i < 2; ++i)
            afrag[i] = load_frag_a(As + (wm * 32 + i * 16) * 32, 32, lane);
#pragma unroll
        for (int j = 0; j < 4; ++j)
            bfrag[j] = load_frag_b(Bs + (wn * 64 + j * 16) * 32, 32, lane);
#pragma unroll
        for (int i = 0; i < 2; ++i)
#pragma unroll
            for (int j = 0; j < 4; ++j)
                acc[i][j] = __builtin_amdgcn_wmma_f32_16x16x32_f16(
                    false, afrag[i], false, bfrag[j], (short)0, acc[i][j], false, false);
    }

    // store: C layout — VGPR v: lane<16 -> M=v, lane>=16 -> M=v+8; N = lane&15
#pragma unroll
    for (int i = 0; i < 2; ++i)
#pragma unroll
        for (int j = 0; j < 4; ++j) {
            const int c = wn * 64 + j * 16 + (lane & 15);
            const int rbase = block_m + wm * 32 + i * 16 + ((lane >> 4) << 3);
#pragma unroll
            for (int v = 0; v < 8; ++v) {
                const int r = rbase + v;
                if (r < M) t1[(size_t)r * 128 + c] = acc[i][j][v];
            }
        }
}

// ---------------------------------------------------------------------------
// GEMM2: t2[M][40] = hh[M][128] @ W2[128][40] (via w2t [48][128] f16).
__global__ __launch_bounds__(256) void gemm2_wmma(const _Float16* __restrict__ hh,
                                                  const _Float16* __restrict__ w2t,
                                                  float* __restrict__ t2, int M) {
    __shared__ __align__(16) _Float16 As2[128 * 128]; // 32 KB
    __shared__ __align__(16) _Float16 Bs2[48 * 128];  // 12 KB
    const int tid  = threadIdx.x;
    const int lane = tid & 31;
    const int wave = tid >> 5;
    const int block_m = blockIdx.x * 128;

    // B: 48*128 halves = 768 x 16B chunks
    {
        const float4* s4 = (const float4*)w2t;
        float4* d4 = (float4*)Bs2;
        for (int c = tid; c < 768; c += 256) d4[c] = s4[c];
    }
    // A: 128 rows x 16 chunks (each chunk = 8 halves = 16B, within one row)
    {
        const float4* s4 = (const float4*)hh;   // row stride = 16 chunks
        float4* d4 = (float4*)As2;
        for (int c = tid; c < 128 * 16; c += 256) {
            const int row = c >> 4;
            const int gr = block_m + row;
            float4 v = make_float4(0.f, 0.f, 0.f, 0.f);
            if (gr < M) v = s4[(size_t)gr * 16 + (c & 15)];
            d4[c] = v;
        }
    }
    __syncthreads();

    const int m0 = wave * 16;
    v8f acc[3] = {};
#pragma unroll
    for (int kk = 0; kk < 4; ++kk) {
        v16h a = load_frag_a(As2 + m0 * 128 + kk * 32, 128, lane);
#pragma unroll
        for (int j = 0; j < 3; ++j) {
            v16h b = load_frag_b(Bs2 + (j * 16) * 128 + kk * 32, 128, lane);
            acc[j] = __builtin_amdgcn_wmma_f32_16x16x32_f16(
                false, a, false, b, (short)0, acc[j], false, false);
        }
    }

    const int rbase = block_m + m0 + ((lane >> 4) << 3);
    const int col0  = lane & 15;
#pragma unroll
    for (int j = 0; j < 3; ++j) {
        const int col = j * 16 + col0;
        if (col < 40) {
#pragma unroll
            for (int v = 0; v < 8; ++v) {
                const int r = rbase + v;
                if (r < M) t2[(size_t)r * 40 + col] = acc[j][v];
            }
        }
    }
}

// ---------------------------------------------------------------------------
// Edge scatter, layer 1: hacc[dst] += t1[src] * dinv[src]*dinv[dst] (128 feats).
// One wave per edge, 4 cols per lane.
__global__ __launch_bounds__(256) void scatter1(const int* __restrict__ src,
                                                const int* __restrict__ dst,
                                                const float* __restrict__ dinv,
                                                const float* __restrict__ t1,
                                                float* __restrict__ hacc, int E) {
    const int e = blockIdx.x * 8 + (threadIdx.x >> 5);
    if (e >= E) return;
    const int lane = threadIdx.x & 31;
    const int s = src[e], d = dst[e];
    const float nrm = dinv[s] * dinv[d];
    const float* tp = t1 + (size_t)s * 128;
    float* hp = hacc + (size_t)d * 128;
#pragma unroll
    for (int j = 0; j < 4; ++j) {
        const int c = lane + j * 32;
        atomicAdd(&hp[c], tp[c] * nrm);
    }
}

// Edge scatter, layer 2: 40 feats, 2 cols per lane (second guarded).
__global__ __launch_bounds__(256) void scatter2(const int* __restrict__ src,
                                                const int* __restrict__ dst,
                                                const float* __restrict__ dinv,
                                                const float* __restrict__ t2,
                                                float* __restrict__ lacc, int E) {
    const int e = blockIdx.x * 8 + (threadIdx.x >> 5);
    if (e >= E) return;
    const int lane = threadIdx.x & 31;
    const int s = src[e], d = dst[e];
    const float nrm = dinv[s] * dinv[d];
    const float* tp = t2 + (size_t)s * 40;
    float* lp = lacc + (size_t)d * 40;
    atomicAdd(&lp[lane], tp[lane] * nrm);
    const int c1 = lane + 32;
    if (c1 < 40) atomicAdd(&lp[c1], tp[c1] * nrm);
}

// Layer-1 epilogue: h = relu(hacc + t1*dinv^2 (self loop) + b1); emit f16 for GEMM2.
// Vectorized: one float4 chunk (4 feats) per thread.
__global__ void epi1(const float* __restrict__ t1, const float* __restrict__ hacc,
                     const float* __restrict__ dinv, const float* __restrict__ b1,
                     _Float16* __restrict__ hh, int M) {
    const size_t gid = (size_t)blockIdx.x * 256 + threadIdx.x;   // chunk id
    if (gid >= (size_t)M * 32) return;                           // 32 chunks / row
    const int i  = (int)(gid >> 5);
    const int c4 = (int)(gid & 31);
    const float di = dinv[i];
    const float dd = di * di;
    const float4 a = ((const float4*)hacc)[gid];
    const float4 t = ((const float4*)t1)[gid];
    const float4 bb = ((const float4*)b1)[c4];
    __align__(8) _Float16 o[4];
    o[0] = (_Float16)fmaxf(a.x + t.x * dd + bb.x, 0.f);
    o[1] = (_Float16)fmaxf(a.y + t.y * dd + bb.y, 0.f);
    o[2] = (_Float16)fmaxf(a.z + t.z * dd + bb.z, 0.f);
    o[3] = (_Float16)fmaxf(a.w + t.w * dd + bb.w, 0.f);
    ((float2*)hh)[gid] = *(const float2*)o;
}

// Layer-2 epilogue + log_softmax. One wave per node (wave32 shuffle reductions).
// out[0 : M*40)        = log_softmax(logits)
// out[M*40 : 2*M*40)   = logits
__global__ __launch_bounds__(256) void epi2(const float* __restrict__ t2,
                                            const float* __restrict__ lacc,
                                            const float* __restrict__ dinv,
                                            const float* __restrict__ b2,
                                            float* __restrict__ out, int M) {
    const int i = blockIdx.x * 8 + (threadIdx.x >> 5);
    if (i >= M) return;
    const int lane = threadIdx.x & 31;
    const float di = dinv[i];
    const float dd = di * di;
    const int c0 = lane;            // always < 40
    const int c1 = lane + 32;       // valid if < 40
    const size_t base = (size_t)i * 40;
    const float l0 = lacc[base + c0] + t2[base + c0] * dd + b2[c0];
    float l1 = -INFINITY;
    if (c1 < 40) l1 = lacc[base + c1] + t2[base + c1] * dd + b2[c1];

    float m = fmaxf(l0, l1);
#pragma unroll
    for (int off = 16; off > 0; off >>= 1) m = fmaxf(m, __shfl_xor(m, off, 32));
    float s = expf(l0 - m) + ((c1 < 40) ? expf(l1 - m) : 0.f);
#pragma unroll
    for (int off = 16; off > 0; off >>= 1) s += __shfl_xor(s, off, 32);
    const float ls = logf(s) + m;

    const size_t off2 = (size_t)M * 40;
    out[base + c0] = l0 - ls;
    out[off2 + base + c0] = l0;
    if (c1 < 40) { out[base + c1] = l1 - ls; out[off2 + base + c1] = l1; }
}

// ---------------------------------------------------------------------------
extern "C" void kernel_launch(void* const* d_in, const int* in_sizes, int n_in,
                              void* d_out, int out_size, void* d_ws, size_t ws_size,
                              hipStream_t stream) {
    const float* x  = (const float*)d_in[0];
    const int*   ei = (const int*)d_in[1];   // [2][E]: row0=src, row1=dst
    const float* W1 = (const float*)d_in[2];
    const float* b1 = (const float*)d_in[3];
    const float* W2 = (const float*)d_in[4];
    const float* b2 = (const float*)d_in[5];
    float* out = (float*)d_out;

    const int H = in_sizes[3];            // 128
    const int F = in_sizes[2] / H;        // 500
    const int M = in_sizes[0] / F;        // 100000 nodes
    const int E = in_sizes[1] / 2;        // 1.6M edges
    (void)n_in; (void)out_size; (void)ws_size;

    const int* srcIdx = ei;
    const int* dstIdx = ei + E;

    // workspace carve-out (512B aligned)
    char* ws = (char*)d_ws;
    size_t off = 0;
    auto take = [&](size_t bytes) -> char* {
        char* p = ws + off;
        off = (off + bytes + 511) & ~(size_t)511;
        return p;
    };
    float*     dinv = (float*)take((size_t)M * 4);
    _Float16*  w1t  = (_Float16*)take(128 * 512 * 2);
    _Float16*  w2t  = (_Float16*)take(48 * 128 * 2);
    float*     t1   = (float*)take((size_t)M * 128 * 4);
    float*     hacc = (float*)take((size_t)M * 128 * 4);
    _Float16*  hh   = (_Float16*)take((size_t)M * 128 * 2);
    // region reuse after epi1 consumes t1/hacc:
    float* t2   = hacc;   // [M][40] fp32
    float* lacc = t1;     // [M][40] fp32 accumulator

    const int T = 256;
    auto nblk = [](size_t n, int t) { return (unsigned)((n + t - 1) / t); };

    // 1) degrees -> dinv (self loop => init 1.0)
    fill_f32<<<nblk(M, T), T, 0, stream>>>(dinv, 1.0f, (size_t)M);
    deg_accum<<<nblk(E, T), T, 0, stream>>>(dstIdx, dinv, E);
    deg_to_dinv<<<nblk(M, T), T, 0, stream>>>(dinv, M);

    // 2) weight prep (transpose + f16)
    prep_w1t<<<nblk(128 * 512, T), T, 0, stream>>>(W1, w1t);
    prep_w2t<<<nblk(48 * 128, T), T, 0, stream>>>(W2, w2t);

    // 3) GEMM1 on WMMA
    gemm1_wmma<<<nblk(M, 128), T, 0, stream>>>(x, w1t, t1, M);

    // 4) layer-1 aggregation
    fill_f32<<<nblk((size_t)M * 128, T), T, 0, stream>>>(hacc, 0.0f, (size_t)M * 128);
    scatter1<<<nblk(E, 8), T, 0, stream>>>(srcIdx, dstIdx, dinv, t1, hacc, E);
    epi1<<<nblk((size_t)M * 32, T), T, 0, stream>>>(t1, hacc, dinv, b1, hh, M);

    // 5) GEMM2 on WMMA (t2 aliases hacc — safe, epi1 already consumed it)
    gemm2_wmma<<<nblk(M, 128), T, 0, stream>>>(hh, w2t, t2, M);

    // 6) layer-2 aggregation (lacc aliases t1 — safe after epi1)
    fill_f32<<<nblk((size_t)M * 40, T), T, 0, stream>>>(lacc, 0.0f, (size_t)M * 40);
    scatter2<<<nblk(E, 8), T, 0, stream>>>(srcIdx, dstIdx, dinv, t2, lacc, E);

    // 7) epilogue + log_softmax -> d_out (logp, then logits)
    epi2<<<nblk(M, 8), T, 0, stream>>>(t2, lacc, dinv, b2, out, M);
}
